// TransformerBlock_33741263078134
// MI455X (gfx1250) — matrix operand order repaired and checked by reading, not compiled, not run on hardware
//
#include <hip/hip_runtime.h>
#include <hip/hip_bf16.h>

// Problem constants (match reference)
#define BB   2
#define TT   2048
#define DD   1024
#define HH   16
#define HDD  64
#define MM   (BB * TT)          // 4096 rows
#define EPSV 1e-5f

typedef __attribute__((ext_vector_type(16))) __bf16 v16bf;
typedef __attribute__((ext_vector_type(8)))  float  v8f;

union FragAB { v16bf v; uint4 q[2]; };

// ---------------------------------------------------------------------------
// CDNA5 async global->LDS staging via inline asm (bypasses builtin AS typing).
// VDST = per-lane LDS byte offset (generic LDS ptr truncated to 32 bits =
// workgroup-relative offset per the aperture encoding); VADDR = 64-bit global
// address pair. Tracked by ASYNCcnt; paired with s_wait_asynccnt below.
// ---------------------------------------------------------------------------
__device__ __forceinline__ void async_b128(const __bf16* g, __bf16* l) {
    unsigned lds_off = (unsigned)(uintptr_t)l;
    unsigned long long ga = (unsigned long long)(uintptr_t)g;
    asm volatile("global_load_async_to_lds_b128 %0, %1, off"
                 :: "v"(lds_off), "v"(ga) : "memory");
}
#define WAIT_ASYNC_1() asm volatile("s_wait_asynccnt 0x1" ::: "memory")
#define WAIT_ASYNC_0() asm volatile("s_wait_asynccnt 0x0" ::: "memory")

__device__ __forceinline__ __bf16 f2bf(float f) {
    unsigned u = __float_as_uint(f);
    unsigned r = (u + 0x7FFFu + ((u >> 16) & 1u)) >> 16;   // RNE
    unsigned short s = (unsigned short)r;
    __bf16 b;
    __builtin_memcpy(&b, &s, 2);
    return b;
}

// ---------------------------------------------------------------------------
// Weight convert + transpose: Wt[n][k] = bf16(W[k][n])
// ---------------------------------------------------------------------------
__global__ __launch_bounds__(256)
void wtrans_kernel(const float* __restrict__ W, __bf16* __restrict__ Wt,
                   int K, int N) {
    int id = blockIdx.x * 256 + threadIdx.x;
    if (id >= K * N) return;
    int n = id / K;
    int k = id - n * K;
    Wt[id] = f2bf(W[(size_t)k * N + n]);
}

// ---------------------------------------------------------------------------
// V transpose: Vt[b][h][d][t] = qkv[b*T+t][2048 + h*64 + d]   (bf16 -> bf16)
// ---------------------------------------------------------------------------
__global__ __launch_bounds__(256)
void vtrans_kernel(const __bf16* __restrict__ qkv, __bf16* __restrict__ vt) {
    int id = blockIdx.x * 256 + threadIdx.x;          // over B*H*HD*T = 4.19M
    int t = id & (TT - 1);
    int d = (id >> 11) & (HDD - 1);
    int h = (id >> 17) & (HH - 1);
    int b = id >> 21;
    vt[id] = qkv[((size_t)(b * TT + t)) * (3 * DD) + 2 * DD + h * HDD + d];
}

// ---------------------------------------------------------------------------
// LayerNorm (fp32 in, bf16 out). One block per row of D=1024, 256 threads.
// ---------------------------------------------------------------------------
__global__ __launch_bounds__(256)
void layernorm_kernel(const float* __restrict__ x, const float* __restrict__ g,
                      const float* __restrict__ bta, __bf16* __restrict__ out) {
    __shared__ float s_sum[256];
    __shared__ float s_sq[256];
    const int row = blockIdx.x;
    const int tid = threadIdx.x;
    const float4 v = ((const float4*)(x + (size_t)row * DD))[tid];
    s_sum[tid] = v.x + v.y + v.z + v.w;
    s_sq[tid]  = v.x * v.x + v.y * v.y + v.z * v.z + v.w * v.w;
    __syncthreads();
    for (int st = 128; st > 0; st >>= 1) {
        if (tid < st) { s_sum[tid] += s_sum[tid + st]; s_sq[tid] += s_sq[tid + st]; }
        __syncthreads();
    }
    const float mu   = s_sum[0] * (1.0f / DD);
    const float var  = s_sq[0] * (1.0f / DD) - mu * mu;
    const float rstd = rsqrtf(var + EPSV);
    const int i = tid * 4;
    __bf16* o = out + (size_t)row * DD + i;
    o[0] = f2bf((v.x - mu) * rstd * g[i + 0] + bta[i + 0]);
    o[1] = f2bf((v.y - mu) * rstd * g[i + 1] + bta[i + 1]);
    o[2] = f2bf((v.z - mu) * rstd * g[i + 2] + bta[i + 2]);
    o[3] = f2bf((v.w - mu) * rstd * g[i + 3] + bta[i + 3]);
}

// ---------------------------------------------------------------------------
// bf16 WMMA GEMM:  C[M,N] = A[M,K] @ Bt[N,K]^T  with fused epilogue.
// Block = 256 threads (8 waves); block tile 256(M) x 64(N).
// Wave w owns rows [m0, m0+32) as 2 M-subtiles x 4 N-subtiles = 8 WMMAs/K-step.
// B tile (64 x 32 K, 4KB) is shared by all waves: staged into LDS with
// double-buffered GLOBAL_LOAD_ASYNC_TO_LDS_B128 (one b128 per thread/stage),
// consumed via ds_load_b128 directly in the native B-fragment layout.
// ---------------------------------------------------------------------------
enum { EPI_BF16 = 0, EPI_GELU = 1, EPI_RESID = 2 };

template <int EPI>
__global__ __launch_bounds__(256)
void gemm_bf16_wmma(const __bf16* __restrict__ A, const __bf16* __restrict__ Bt,
                    const float* __restrict__ resid, void* __restrict__ out,
                    int M, int N, int K) {
    __shared__ __bf16 Blds[2][64][32];          // [buf][n][k]  (8 KB)

    const int tid  = threadIdx.x;
    const int wave = tid >> 5;
    const int lane = tid & 31;
    const int lm   = lane & 15;
    const int hi   = (lane >> 4) & 1;

    const int m0 = blockIdx.x * 256 + wave * 32;
    const int n0 = blockIdx.y * 64;

    // cooperative B staging: thread -> (col n, 8-element K chunk)
    const int bn = tid >> 2;
    const int bk = (tid & 3) * 8;
    const __bf16* bsrc = Bt + (size_t)(n0 + bn) * K + bk;

    const v8f zero = {0.f, 0.f, 0.f, 0.f, 0.f, 0.f, 0.f, 0.f};
    v8f acc[2][4];
#pragma unroll
    for (int i = 0; i < 2; ++i)
#pragma unroll
        for (int j = 0; j < 4; ++j) acc[i][j] = zero;

    // A-frag per lane: row, K elems {klo..klo+7, klo+16..klo+23}, klo = 8*hi
    const __bf16* arow0 = A + (size_t)(m0 + lm) * K + hi * 8;
    const __bf16* arow1 = arow0 + (size_t)16 * K;

    const int nk = K >> 5;
    async_b128(bsrc, &Blds[0][bn][bk]);                      // stage 0
    for (int c = 0; c < nk; ++c) {
        const int k0  = c << 5;
        const int cur = c & 1;
        if (c + 1 < nk) {
            async_b128(bsrc + (k0 + 32), &Blds[cur ^ 1][bn][bk]);
            WAIT_ASYNC_1();                                  // stage c landed
        } else {
            WAIT_ASYNC_0();
        }
        __syncthreads();

        FragAB a0, a1;
        { const uint4* p = (const uint4*)(arow0 + k0); a0.q[0] = p[0]; a0.q[1] = p[2]; }
        { const uint4* p = (const uint4*)(arow1 + k0); a1.q[0] = p[0]; a1.q[1] = p[2]; }
#pragma unroll
        for (int j = 0; j < 4; ++j) {
            FragAB b;    // B-frag: lane -> col j*16+lm, K {8*hi .. 8*hi+7, 8*hi+16 .. 8*hi+23}
            b.q[0] = *(const uint4*)&Blds[cur][j * 16 + lm][hi * 8];
            b.q[1] = *(const uint4*)&Blds[cur][j * 16 + lm][hi * 8 + 16];
            acc[0][j] = __builtin_amdgcn_wmma_f32_16x16x32_bf16(
                false, a0.v, false, b.v, (short)0, acc[0][j], false, false);
            acc[1][j] = __builtin_amdgcn_wmma_f32_16x16x32_bf16(
                false, a1.v, false, b.v, (short)0, acc[1][j], false, false);
        }
        __syncthreads();        // protect buffer reuse at stage c+2
    }

    // C layout: VGPR r -> row m0 + 16i + r + 8*hi, col n0 + 16j + lm
#pragma unroll
    for (int i = 0; i < 2; ++i) {
#pragma unroll
        for (int j = 0; j < 4; ++j) {
            const int col = n0 + j * 16 + lm;
#pragma unroll
            for (int r = 0; r < 8; ++r) {
                const int row = m0 + i * 16 + r + hi * 8;
                const float v = acc[i][j][r];
                const size_t idx = (size_t)row * N + col;
                if (EPI == EPI_BF16) {
                    ((__bf16*)out)[idx] = f2bf(v);
                } else if (EPI == EPI_GELU) {
                    const float gl = 0.5f * v * (1.0f + erff(v * 0.70710678118654752f));
                    ((__bf16*)out)[idx] = f2bf(gl);
                } else {  // EPI_RESID
                    ((float*)out)[idx] = resid[idx] + v;
                }
            }
        }
    }
}

// ---------------------------------------------------------------------------
// Flash attention (causal). 128 threads = 4 waves; each wave owns one
// 16-query tile. HD=64. S = Q K^T via 4 WMMAs / 32-key chunk, P V via 4.
// ---------------------------------------------------------------------------
__global__ __launch_bounds__(128)
void attention_wmma(const __bf16* __restrict__ qkv, const __bf16* __restrict__ vt,
                    __bf16* __restrict__ y) {
    __shared__ __bf16 P[4][16][32];

    const int wave = threadIdx.x >> 5;
    const int lane = threadIdx.x & 31;
    const int lm   = lane & 15;
    const int hi   = (lane >> 4) & 1;

    const int q0 = (blockIdx.x * 4 + wave) * 16;
    const int h  = blockIdx.y;
    const int b  = blockIdx.z;

    const __bf16* Qb = qkv + (size_t)b * TT * (3 * DD) + (size_t)h * HDD;
    const __bf16* Kb = Qb + DD;
    const __bf16* Vb = vt + (size_t)(b * HH + h) * HDD * TT;

    // Q fragments for dim-chunks {0..31},{32..63}; held for entire key loop
    FragAB qa[2];
    {
        const __bf16* qrow = Qb + (size_t)(q0 + lm) * (3 * DD) + hi * 8;
#pragma unroll
        for (int t = 0; t < 2; ++t) {
            const uint4* qp = (const uint4*)(qrow + t * 32);
            qa[t].q[0] = qp[0];
            qa[t].q[1] = qp[2];
        }
    }

    const v8f zero = {0.f, 0.f, 0.f, 0.f, 0.f, 0.f, 0.f, 0.f};
    v8f o[4];
#pragma unroll
    for (int j = 0; j < 4; ++j) o[j] = zero;
    float rmax[8], rsum[8];
#pragma unroll
    for (int r = 0; r < 8; ++r) { rmax[r] = -1e30f; rsum[r] = 0.f; }

    const int nch = (q0 + 16 + 31) >> 5;           // causal: keys <= q0+15
    for (int c = 0; c < nch; ++c) {
        const int kc0 = c * 32;
        v8f s[2] = {zero, zero};
#pragma unroll
        for (int t = 0; t < 2; ++t) {
            const __bf16* krow = Kb + (size_t)(kc0 + t * 16 + lm) * (3 * DD) + hi * 8;
#pragma unroll
            for (int kk = 0; kk < 2; ++kk) {
                FragAB kb;
                const uint4* kp = (const uint4*)(krow + kk * 32);
                kb.q[0] = kp[0];
                kb.q[1] = kp[2];
                s[t] = __builtin_amdgcn_wmma_f32_16x16x32_bf16(
                    false, qa[kk].v, false, kb.v, (short)0, s[t], false, false);
            }
        }
        // online softmax per row (rows split by lane half; reduce over 16 lanes)
#pragma unroll
        for (int r = 0; r < 8; ++r) {
            const int row = q0 + r + hi * 8;
            float s0 = s[0][r] * 0.125f;
            float s1 = s[1][r] * 0.125f;
            if (kc0 + lm > row)      s0 = -1e30f;
            if (kc0 + 16 + lm > row) s1 = -1e30f;
            float mx = fmaxf(s0, s1);
#pragma unroll
            for (int d = 1; d < 16; d <<= 1) mx = fmaxf(mx, __shfl_xor(mx, d, 32));
            const float nm = fmaxf(rmax[r], mx);
            const float sf = __expf(rmax[r] - nm);
            const float p0 = __expf(s0 - nm);
            const float p1 = __expf(s1 - nm);
            float ps = p0 + p1;
#pragma unroll
            for (int d = 1; d < 16; d <<= 1) ps += __shfl_xor(ps, d, 32);
            rsum[r] = rsum[r] * sf + ps;
            rmax[r] = nm;
#pragma unroll
            for (int j = 0; j < 4; ++j) o[j][r] *= sf;
            P[wave][r + hi * 8][lm]      = f2bf(p0);
            P[wave][r + hi * 8][lm + 16] = f2bf(p1);
        }
        // reshape P (C-layout) -> A-layout fragment via LDS (in-order per wave)
        FragAB pa;
        pa.q[0] = *(const uint4*)&P[wave][lm][hi * 8];
        pa.q[1] = *(const uint4*)&P[wave][lm][hi * 8 + 16];
        // O += P @ V  (V pre-transposed: rows = dims, contiguous over keys)
#pragma unroll
        for (int j = 0; j < 4; ++j) {
            FragAB vb;
            const uint4* vp = (const uint4*)(Vb + (size_t)(j * 16 + lm) * TT + kc0 + hi * 8);
            vb.q[0] = vp[0];
            vb.q[1] = vp[2];
            o[j] = __builtin_amdgcn_wmma_f32_16x16x32_bf16(
                false, pa.v, false, vb.v, (short)0, o[j], false, false);
        }
    }

    // normalize and emit bf16 into y[b,t,D] at head slice
#pragma unroll
    for (int j = 0; j < 4; ++j) {
#pragma unroll
        for (int r = 0; r < 8; ++r) {
            const int row = q0 + r + hi * 8;
            const float val = o[j][r] / rsum[r];
            y[((size_t)b * TT + row) * DD + h * HDD + j * 16 + lm] = f2bf(val);
        }
    }
}

// ---------------------------------------------------------------------------
// Host-side orchestration
// ---------------------------------------------------------------------------
extern "C" void kernel_launch(void* const* d_in, const int* in_sizes, int n_in,
                              void* d_out, int out_size, void* d_ws, size_t ws_size,
                              hipStream_t stream) {
    (void)in_sizes; (void)n_in; (void)out_size; (void)ws_size;

    const float* x      = (const float*)d_in[0];
    const float* w_attn = (const float*)d_in[1];
    const float* w_proj = (const float*)d_in[2];
    const float* w_fc   = (const float*)d_in[3];
    const float* w_out  = (const float*)d_in[4];
    const float* ln1_g  = (const float*)d_in[5];
    const float* ln1_b  = (const float*)d_in[6];
    const float* ln2_g  = (const float*)d_in[7];
    const float* ln2_b  = (const float*)d_in[8];
    float* out = (float*)d_out;

    char* ws = (char*)d_ws;
    size_t off = 0;
    auto alloc = [&](size_t bytes) -> char* {
        char* p = ws + off;
        off += (bytes + 255) & ~(size_t)255;
        return p;
    };

    __bf16* wattn_t = (__bf16*)alloc((size_t)3 * DD * DD * 2);     // [3072][1024]
    __bf16* wproj_t = (__bf16*)alloc((size_t)DD * DD * 2);         // [1024][1024]
    __bf16* wfc_t   = (__bf16*)alloc((size_t)4 * DD * DD * 2);     // [4096][1024]
    __bf16* wout_t  = (__bf16*)alloc((size_t)DD * 4 * DD * 2);     // [1024][4096]
    __bf16* xn1     = (__bf16*)alloc((size_t)MM * DD * 2);
    __bf16* qkvb    = (__bf16*)alloc((size_t)MM * 3 * DD * 2);
    __bf16* vtb     = (__bf16*)alloc((size_t)BB * HH * HDD * TT * 2);
    __bf16* yb      = (__bf16*)alloc((size_t)MM * DD * 2);
    float*  x1      = (float*)alloc((size_t)MM * DD * 4);
    __bf16* h2      = (__bf16*)alloc((size_t)MM * DD * 2);
    __bf16* afc     = (__bf16*)alloc((size_t)MM * 4 * DD * 2);

    // 1) weights -> bf16 transposed [N][K]
    wtrans_kernel<<<(DD * 3 * DD + 255) / 256, 256, 0, stream>>>(w_attn, wattn_t, DD, 3 * DD);
    wtrans_kernel<<<(DD * DD + 255) / 256,     256, 0, stream>>>(w_proj, wproj_t, DD, DD);
    wtrans_kernel<<<(DD * 4 * DD + 255) / 256, 256, 0, stream>>>(w_fc,   wfc_t,   DD, 4 * DD);
    wtrans_kernel<<<(4 * DD * DD + 255) / 256, 256, 0, stream>>>(w_out,  wout_t,  4 * DD, DD);

    // 2) LN1 -> bf16
    layernorm_kernel<<<MM, 256, 0, stream>>>(x, ln1_g, ln1_b, xn1);

    // 3) qkv = xn1 @ w_attn  -> bf16 [4096, 3072]
    gemm_bf16_wmma<EPI_BF16><<<dim3(MM / 256, 3 * DD / 64), 256, 0, stream>>>(
        xn1, wattn_t, nullptr, qkvb, MM, 3 * DD, DD);

    // 4) V transpose -> [b,h,d,t]
    vtrans_kernel<<<(BB * HH * HDD * TT) / 256, 256, 0, stream>>>(qkvb, vtb);

    // 5) causal flash attention -> yb bf16 [4096, 1024]
    attention_wmma<<<dim3(TT / 64, HH, BB), 128, 0, stream>>>(qkvb, vtb, yb);

    // 6) x1 = x + yb @ w_proj  (fp32)
    gemm_bf16_wmma<EPI_RESID><<<dim3(MM / 256, DD / 64), 256, 0, stream>>>(
        yb, wproj_t, x, x1, MM, DD, DD);

    // 7) LN2 -> bf16
    layernorm_kernel<<<MM, 256, 0, stream>>>(x1, ln2_g, ln2_b, h2);

    // 8) afc = gelu(h2 @ w_fc) -> bf16 [4096, 4096]
    gemm_bf16_wmma<EPI_GELU><<<dim3(MM / 256, 4 * DD / 64), 256, 0, stream>>>(
        h2, wfc_t, nullptr, afc, MM, 4 * DD, DD);

    // 9) out = x1 + afc @ w_out (fp32)
    gemm_bf16_wmma<EPI_RESID><<<dim3(MM / 256, DD / 64), 256, 0, stream>>>(
        afc, wout_t, x1, out, MM, DD, 4 * DD);
}
